// SINDyAutoencoder_15753940041928
// MI455X (gfx1250) — compile-verified, matrix operand order checked by reading, not run on hardware
//
#include <hip/hip_runtime.h>
#include <hip/hip_bf16.h>

// ---------- types ----------
typedef __attribute__((ext_vector_type(16))) __bf16        v16bf;
typedef __attribute__((ext_vector_type(8)))  float         v8f;
typedef __attribute__((ext_vector_type(4)))  unsigned int  u32x4;
typedef __attribute__((ext_vector_type(4)))  float         f32x4;

#define BF_STRIDE 520   // ushorts per row (1040 B: 16B-aligned rows, conflict-skewed)

__device__ __forceinline__ unsigned short f2bf(float f) {
    unsigned int u = __float_as_uint(f);
    unsigned int r = u + 0x7FFFu + ((u >> 16) & 1u);   // round-to-nearest-even
    return (unsigned short)(r >> 16);
}
__device__ __forceinline__ unsigned int pack2(float a, float b) {
    return (unsigned int)f2bf(a) | ((unsigned int)f2bf(b) << 16);
}
__device__ __forceinline__ float bflo(unsigned int u) { return __uint_as_float(u << 16); }
__device__ __forceinline__ float bfhi(unsigned int u) { return __uint_as_float(u & 0xFFFF0000u); }

union BfFrag { v16bf v; u32x4 q[2]; };

// A fragment (16x32 bf16): lane l: m = mbase + (l&15), half = l>>4
__device__ __forceinline__ v16bf loadA(const unsigned short* sbf, int m, int half, int kbase) {
    BfFrag u;
    const unsigned short* p = sbf + m * BF_STRIDE + kbase + half * 8;
    u.q[0] = *(const u32x4*)(p);
    u.q[1] = *(const u32x4*)(p + 16);
    return u.v;
}
// B fragment from transposed weights wT[N][Kstride] (bf16)
__device__ __forceinline__ v16bf loadB(const unsigned short* __restrict__ wT,
                                       int n, int half, int kbase, int Kstride) {
    BfFrag u;
    const unsigned short* p = wT + (size_t)n * Kstride + kbase + half * 8;
    u.q[0] = *(const u32x4*)(p);
    u.q[1] = *(const u32x4*)(p + 16);
    return u.v;
}

// 64-row GEMM, 512-wide output: each wave owns 4 N-tiles; per k-step one B load
// feeds 4 M-group WMMAs (4x weight reuse from registers). bias+bf16 -> bufOut.
__device__ __forceinline__ void gemm4_to_buf(const unsigned short* bufIn,
                                             const unsigned short* __restrict__ wT,
                                             const float* __restrict__ bias,
                                             unsigned short* bufOut,
                                             int wave, int lane, int K) {
    const int nl = lane & 15, half = lane >> 4;
    const v8f vz = {0.f,0.f,0.f,0.f,0.f,0.f,0.f,0.f};
    for (int t = wave; t < 32; t += 8) {
        const int n0 = t * 16;
        v8f acc0 = vz, acc1 = vz, acc2 = vz, acc3 = vz;
        for (int kb = 0; kb < K; kb += 32) {
            v16bf b  = loadB(wT, n0 + nl, half, kb, K);
            v16bf a0 = loadA(bufIn,  0 + nl, half, kb);
            v16bf a1 = loadA(bufIn, 16 + nl, half, kb);
            v16bf a2 = loadA(bufIn, 32 + nl, half, kb);
            v16bf a3 = loadA(bufIn, 48 + nl, half, kb);
            acc0 = __builtin_amdgcn_wmma_f32_16x16x32_bf16(false, a0, false, b, (short)0, acc0, false, false);
            acc1 = __builtin_amdgcn_wmma_f32_16x16x32_bf16(false, a1, false, b, (short)0, acc1, false, false);
            acc2 = __builtin_amdgcn_wmma_f32_16x16x32_bf16(false, a2, false, b, (short)0, acc2, false, false);
            acc3 = __builtin_amdgcn_wmma_f32_16x16x32_bf16(false, a3, false, b, (short)0, acc3, false, false);
        }
        const int n = n0 + nl;
        const float bz = bias[n];
        #pragma unroll
        for (int r = 0; r < 8; ++r) {
            bufOut[( 0 + r + 8 * half) * BF_STRIDE + n] = f2bf(acc0[r] + bz);
            bufOut[(16 + r + 8 * half) * BF_STRIDE + n] = f2bf(acc1[r] + bz);
            bufOut[(32 + r + 8 * half) * BF_STRIDE + n] = f2bf(acc2[r] + bz);
            bufOut[(48 + r + 8 * half) * BF_STRIDE + n] = f2bf(acc3[r] + bz);
        }
    }
}

// LayerNorm(512) + SiLU in place on 64-row bf16 buffer; f32 stats. 256 threads.
__device__ __forceinline__ void ln_silu64(unsigned short* buf, float* sRed, float* sStat,
                                          const float* __restrict__ g,
                                          const float* __restrict__ be, int tid) {
    const int row = tid >> 2, part = tid & 3;      // 64 rows x 4 parts x 128 cols
    unsigned short* prow = buf + row * BF_STRIDE + part * 128;
    float s1 = 0.f, s2 = 0.f;
    for (int i = 0; i < 16; ++i) {
        u32x4 q = *(const u32x4*)(prow + i * 8);
        #pragma unroll
        for (int j = 0; j < 4; ++j) {
            float a = bflo(q[j]), b = bfhi(q[j]);
            s1 += a + b; s2 += a * a + b * b;
        }
    }
    sRed[(row * 4 + part) * 2 + 0] = s1;
    sRed[(row * 4 + part) * 2 + 1] = s2;
    __syncthreads();
    if (tid < 64) {
        float a = 0.f, b = 0.f;
        #pragma unroll
        for (int p = 0; p < 4; ++p) {
            a += sRed[(tid * 4 + p) * 2 + 0];
            b += sRed[(tid * 4 + p) * 2 + 1];
        }
        const float mean = a * (1.0f / 512.0f);
        const float var  = b * (1.0f / 512.0f) - mean * mean;
        sStat[tid * 2 + 0] = mean;
        sStat[tid * 2 + 1] = rsqrtf(var + 1e-5f);
    }
    __syncthreads();
    const float mean = sStat[row * 2 + 0], rstd = sStat[row * 2 + 1];
    const float* gp = g + part * 128;
    const float* bp = be + part * 128;
    for (int i = 0; i < 16; ++i) {
        u32x4 q = *(const u32x4*)(prow + i * 8);
        f32x4 g0 = *(const f32x4*)(gp + i * 8);
        f32x4 g1 = *(const f32x4*)(gp + i * 8 + 4);
        f32x4 b0 = *(const f32x4*)(bp + i * 8);
        f32x4 b1 = *(const f32x4*)(bp + i * 8 + 4);
        u32x4 o;
        #pragma unroll
        for (int j = 0; j < 4; ++j) {
            float ga = (j < 2) ? g0[j * 2]     : g1[(j - 2) * 2];
            float gb = (j < 2) ? g0[j * 2 + 1] : g1[(j - 2) * 2 + 1];
            float ba = (j < 2) ? b0[j * 2]     : b1[(j - 2) * 2];
            float bb = (j < 2) ? b0[j * 2 + 1] : b1[(j - 2) * 2 + 1];
            float a = (bflo(q[j]) - mean) * rstd * ga + ba;
            float c = (bfhi(q[j]) - mean) * rstd * gb + bb;
            a = a / (1.0f + __expf(-a));
            c = c / (1.0f + __expf(-c));
            o[j] = pack2(a, c);
        }
        *(u32x4*)(prow + i * 8) = o;
    }
    __syncthreads();
}

// ---------- weight prep: f32 [K,N] -> bf16 transposed+padded [Npad][Kpad] ----------
__global__ void transpose_bf16(const float* __restrict__ src, unsigned short* __restrict__ dst,
                               int K, int N, int Kpad, int Npad) {
    int idx = blockIdx.x * blockDim.x + threadIdx.x;
    if (idx >= Kpad * Npad) return;
    int n = idx / Kpad, k = idx % Kpad;
    float v = (k < K && n < N) ? src[(size_t)k * N + n] : 0.0f;
    dst[(size_t)n * Kpad + k] = f2bf(v);
}

// ---------- fully fused SINDy autoencoder: 64 rows per block ----------
__global__ __launch_bounds__(256) void sindy_fused(
    const float* __restrict__ x0, const float* __restrict__ dtn,
    const float* __restrict__ eb1, const float* __restrict__ eg1, const float* __restrict__ ee1,
    const float* __restrict__ eb2, const float* __restrict__ eg2, const float* __restrict__ ee2,
    const float* __restrict__ eb3,
    const float* __restrict__ db1, const float* __restrict__ dg1, const float* __restrict__ de1,
    const float* __restrict__ db2, const float* __restrict__ dg2, const float* __restrict__ de2,
    const float* __restrict__ db3, const float* __restrict__ Xi,
    const unsigned short* __restrict__ w1T, const unsigned short* __restrict__ w2T,
    const unsigned short* __restrict__ w3T, const unsigned short* __restrict__ d1T,
    const unsigned short* __restrict__ d2T, const unsigned short* __restrict__ d3T,
    float* __restrict__ out) {

    __shared__ __attribute__((aligned(16))) unsigned short bufA[64 * BF_STRIDE];
    __shared__ __attribute__((aligned(16))) unsigned short bufB[64 * BF_STRIDE];
    __shared__ float sRed[64 * 4 * 2];
    __shared__ float sStat[64 * 2];
    __shared__ float sXi[45 * 8];
    __shared__ float sZ[64 * 8];

    const int tid  = threadIdx.x;
    const int lane = tid & 31;
    const int wave = tid >> 5;
    const int nl   = lane & 15;
    const int half = lane >> 4;
    const int rowBase = blockIdx.x * 64;
    const v8f vz = {0.f,0.f,0.f,0.f,0.f,0.f,0.f,0.f};

    // stage X tile (64x64) as bf16 into bufA, Xi (45x8) f32 into LDS
    {
        const int r = tid >> 2, c16 = (tid & 3) * 16;
        const float* px = x0 + (size_t)(rowBase + r) * 64 + c16;
        u32x4 o0, o1;
        #pragma unroll
        for (int v = 0; v < 4; ++v) {
            f32x4 xv = *(const f32x4*)(px + v * 4);
            unsigned int p0 = pack2(xv[0], xv[1]);
            unsigned int p1 = pack2(xv[2], xv[3]);
            if (v < 2) { o0[v * 2] = p0; o0[v * 2 + 1] = p1; }
            else       { o1[(v - 2) * 2] = p0; o1[(v - 2) * 2 + 1] = p1; }
        }
        *(u32x4*)(bufA + r * BF_STRIDE + c16)     = o0;
        *(u32x4*)(bufA + r * BF_STRIDE + c16 + 8) = o1;
        for (int i = tid; i < 360; i += 256) sXi[i] = Xi[i];
    }
    __syncthreads();

    // enc L1: bufA[64,64] @ [64,512] -> bufB ; LN+SiLU in place
    gemm4_to_buf(bufA, w1T, eb1, bufB, wave, lane, 64);
    __syncthreads();
    ln_silu64(bufB, sRed, sStat, eg1, ee1, tid);

    // enc L2: bufB @ [512,512] -> bufA ; LN+SiLU
    gemm4_to_buf(bufB, w2T, eb2, bufA, wave, lane, 512);
    __syncthreads();
    ln_silu64(bufA, sRed, sStat, eg2, ee2, tid);

    // enc L3 (N 8->16 pad): waves 0..3 each take one M-group; tanh -> sZ
    if (wave < 4) {
        const int mg = wave;
        v8f acc = vz;
        for (int kb = 0; kb < 512; kb += 32) {
            v16bf a = loadA(bufA, mg * 16 + nl, half, kb);
            v16bf b = loadB(w3T, nl, half, kb, 512);
            acc = __builtin_amdgcn_wmma_f32_16x16x32_bf16(false, a, false, b, (short)0, acc, false, false);
        }
        if (nl < 8) {
            const float bz = eb3[nl];
            #pragma unroll
            for (int r = 0; r < 8; ++r)
                sZ[(mg * 16 + r + 8 * half) * 8 + nl] = tanhf(acc[r] + bz);
        }
    }
    __syncthreads();

    // 20-step Euler: dz/dtau = Theta(z) @ Xi ; one row per lane (64 lanes), f32
    if (tid < 64) {
        float z[8];
        #pragma unroll
        for (int j = 0; j < 8; ++j) z[j] = sZ[tid * 8 + j];
        const float dt = dtn[rowBase + tid] * (1.0f / 20.0f);
        for (int s = 0; s < 20; ++s) {
            float zd[8];
            #pragma unroll
            for (int j = 0; j < 8; ++j) zd[j] = sXi[j];                  // const term
            #pragma unroll
            for (int i = 0; i < 8; ++i) {
                const float zi = z[i];
                #pragma unroll
                for (int j = 0; j < 8; ++j) zd[j] += zi * sXi[(1 + i) * 8 + j];
            }
            int rI = 9;                                                   // triu quadratic
            #pragma unroll
            for (int i = 0; i < 8; ++i)
                for (int k = i; k < 8; ++k, ++rI) {
                    const float p = z[i] * z[k];
                    #pragma unroll
                    for (int j = 0; j < 8; ++j) zd[j] += p * sXi[rI * 8 + j];
                }
            #pragma unroll
            for (int j = 0; j < 8; ++j) z[j] += zd[j] * dt;
        }
        // z -> bufB rows, K padded 8->32 with zeros (16B-vector stores)
        u32x4 zq, zero = {0u, 0u, 0u, 0u};
        #pragma unroll
        for (int j = 0; j < 4; ++j) zq[j] = pack2(z[2 * j], z[2 * j + 1]);
        *(u32x4*)(bufB + tid * BF_STRIDE)      = zq;
        *(u32x4*)(bufB + tid * BF_STRIDE + 8)  = zero;
        *(u32x4*)(bufB + tid * BF_STRIDE + 16) = zero;
        *(u32x4*)(bufB + tid * BF_STRIDE + 24) = zero;
    }
    __syncthreads();

    // dec L1: bufB[64,32] @ [32,512] -> bufA ; LN+SiLU
    gemm4_to_buf(bufB, d1T, db1, bufA, wave, lane, 32);
    __syncthreads();
    ln_silu64(bufA, sRed, sStat, dg1, de1, tid);

    // dec L2: bufA @ [512,512] -> bufB ; LN+SiLU
    gemm4_to_buf(bufA, d2T, db2, bufB, wave, lane, 512);
    __syncthreads();
    ln_silu64(bufB, sRed, sStat, dg2, de2, tid);

    // dec L3: bufB[64,512] @ [512,64] + bias -> out ; 16 tiles, B reused over M-pairs
    {
        const int nt = wave & 3;
        const int mg = (wave >> 2) * 2;           // 0 or 2; handles mg and mg+1
        const int n0 = nt * 16;
        v8f acc0 = vz, acc1 = vz;
        for (int kb = 0; kb < 512; kb += 32) {
            v16bf b  = loadB(d3T, n0 + nl, half, kb, 512);
            v16bf a0 = loadA(bufB, (mg    ) * 16 + nl, half, kb);
            v16bf a1 = loadA(bufB, (mg + 1) * 16 + nl, half, kb);
            acc0 = __builtin_amdgcn_wmma_f32_16x16x32_bf16(false, a0, false, b, (short)0, acc0, false, false);
            acc1 = __builtin_amdgcn_wmma_f32_16x16x32_bf16(false, a1, false, b, (short)0, acc1, false, false);
        }
        const int n = n0 + nl;
        const float bz = db3[n];
        #pragma unroll
        for (int r = 0; r < 8; ++r) {
            out[(size_t)(rowBase + (mg    ) * 16 + r + 8 * half) * 64 + n] = acc0[r] + bz;
            out[(size_t)(rowBase + (mg + 1) * 16 + r + 8 * half) * 64 + n] = acc1[r] + bz;
        }
    }
}

extern "C" void kernel_launch(void* const* d_in, const int* in_sizes, int n_in,
                              void* d_out, int out_size, void* d_ws, size_t ws_size,
                              hipStream_t stream) {
    const float* x0  = (const float*)d_in[0];
    const float* dtn = (const float*)d_in[1];
    const float* eW1 = (const float*)d_in[4];
    const float* eb1 = (const float*)d_in[5];
    const float* eg1 = (const float*)d_in[6];
    const float* ee1 = (const float*)d_in[7];
    const float* eW2 = (const float*)d_in[8];
    const float* eb2 = (const float*)d_in[9];
    const float* eg2 = (const float*)d_in[10];
    const float* ee2 = (const float*)d_in[11];
    const float* eW3 = (const float*)d_in[12];
    const float* eb3 = (const float*)d_in[13];
    const float* dW1 = (const float*)d_in[14];
    const float* db1 = (const float*)d_in[15];
    const float* dg1 = (const float*)d_in[16];
    const float* de1 = (const float*)d_in[17];
    const float* dW2 = (const float*)d_in[18];
    const float* db2 = (const float*)d_in[19];
    const float* dg2 = (const float*)d_in[20];
    const float* de2 = (const float*)d_in[21];
    const float* dW3 = (const float*)d_in[22];
    const float* db3 = (const float*)d_in[23];
    const float* Xi  = (const float*)d_in[24];
    float* out = (float*)d_out;

    // ws layout (ushort elements): transposed bf16 weights
    unsigned short* ws  = (unsigned short*)d_ws;
    unsigned short* w1T = ws;                 // [512][64]
    unsigned short* w2T = w1T + 512 * 64;     // [512][512]
    unsigned short* w3T = w2T + 512 * 512;    // [16][512]  (N 8->16 pad)
    unsigned short* d1T = w3T + 16 * 512;     // [512][32]  (K 8->32 pad)
    unsigned short* d2T = d1T + 512 * 32;     // [512][512]
    unsigned short* d3T = d2T + 512 * 512;    // [64][512]

    auto prep = [&](const float* src, unsigned short* dst, int K, int N, int Kp, int Np) {
        int total = Kp * Np;
        transpose_bf16<<<(total + 255) / 256, 256, 0, stream>>>(src, dst, K, N, Kp, Np);
    };
    prep(eW1, w1T,  64, 512,  64, 512);
    prep(eW2, w2T, 512, 512, 512, 512);
    prep(eW3, w3T, 512,   8, 512,  16);
    prep(dW1, d1T,   8, 512,  32, 512);
    prep(dW2, d2T, 512, 512, 512, 512);
    prep(dW3, d3T, 512,  64, 512,  64);

    const int B = in_sizes[0] / 64;           // 131072
    sindy_fused<<<B / 64, 256, 0, stream>>>(
        x0, dtn,
        eb1, eg1, ee1, eb2, eg2, ee2, eb3,
        db1, dg1, de1, db2, dg2, de2, db3, Xi,
        w1T, w2T, w3T, d1T, d2T, d3T, out);
}
